// ChatGLMAttention_1958505087525
// MI455X (gfx1250) — matrix operand order repaired and checked by reading, not compiled, not run on hardware
//
#include <hip/hip_runtime.h>
#include <hip/hip_bf16.h>

// ChatGLM attention block for MI455X (gfx1250), wave32 + WMMA bf16.
// Pipeline: f32->bf16 convert, QKV GEMM + bias + 2D RoPE (V stored transposed),
// flash attention (v_wmma for QK^T and P*V), output projection GEMM.
// GEMMs use double-buffered LDS stages (K=64) fed by gfx1250 async
// global->LDS copies when the toolchain exposes them.

typedef __bf16 bf16;
typedef __attribute__((ext_vector_type(16))) __bf16 v16bf;
typedef __attribute__((ext_vector_type(8)))  float  v8f;
typedef __attribute__((ext_vector_type(4)))  int    v4i;

#define S_LEN 2048
#define NHEAD 32
#define HDIM  128
#define HID   4096
#define QKVN  12288

#define KSTEP 64
#define LDP   72   // padded LDS row (64 + 8 bf16) -> conflict-free frag reads

// ---------------------------------------------------- async global->LDS ------
#if defined(__has_builtin)
#if __has_builtin(__builtin_amdgcn_global_load_async_to_lds_b128)
#define HAVE_ASYNC_LDS 1
#endif
#endif

#ifdef HAVE_ASYNC_LDS
typedef __attribute__((address_space(1))) v4i* g_v4i_p;  // global int4*
typedef __attribute__((address_space(3))) v4i* l_v4i_p;  // LDS int4*

__device__ __forceinline__ void cp16_g2l(const bf16* g, bf16* l) {
  // Launder through integers: generic/global share addresses; generic-LDS
  // low 32 bits are the LDS byte offset (ISA 10.2 aperture mapping).
  g_v4i_p gp = (g_v4i_p)(unsigned long long)g;
  l_v4i_p lp = (l_v4i_p)(unsigned int)(unsigned long long)l;
  __builtin_amdgcn_global_load_async_to_lds_b128(gp, lp, 0, 0);
}
__device__ __forceinline__ void async_wait0() {
#if __has_builtin(__builtin_amdgcn_s_wait_asynccnt)
  __builtin_amdgcn_s_wait_asynccnt(0);
#else
  asm volatile("s_wait_asynccnt 0x0" ::: "memory");
#endif
}
#else
__device__ __forceinline__ void cp16_g2l(const bf16* g, bf16* l) {
  *(uint4*)l = *(const uint4*)g;
}
__device__ __forceinline__ void async_wait0() {}
#endif

union Frag { uint4 u[2]; v16bf v; };

// A-matrix 16x32 bf16 fragment from row-major block, leading dim ld.
// Lane m=lane&15 (row), half=lane>>4. Elem e -> K=half*16+e.
__device__ __forceinline__ v16bf frag_a(const bf16* p, int ld) {
  const int lane = threadIdx.x & 31;
  const int m = lane & 15, h = lane >> 4;
  const bf16* r = p + (size_t)m * ld;
  Frag f;
  f.u[0] = *(const uint4*)(r + h * 16);
  f.u[1] = *(const uint4*)(r + h * 16 + 8);
  return f.v;
}

// B-matrix 32x16 bf16 fragment. B[k][n] = p[n*ld + k] (rows of p are the N dim,
// contiguous along K). Lane n=lane&15 (col), half=lane>>4. Elem e -> K=half*16+e.
__device__ __forceinline__ v16bf frag_b(const bf16* p, int ld) {
  const int lane = threadIdx.x & 31;
  const int n = lane & 15, h = lane >> 4;
  const bf16* r = p + (size_t)n * ld + h * 16;
  Frag f;
  f.u[0] = *(const uint4*)r;
  f.u[1] = *(const uint4*)(r + 8);
  return f.v;
}

__device__ __forceinline__ v8f wmma_bf16(v16bf a, v16bf b, v8f c) {
  return __builtin_amdgcn_wmma_f32_16x16x32_bf16(false, a, false, b, (short)0, c,
                                                 false, false);
}

__device__ __forceinline__ v8f vzero8() {
  v8f z;
  #pragma unroll
  for (int i = 0; i < 8; ++i) z[i] = 0.0f;
  return z;
}

// ------------------------------------------- shared 128x128 GEMM main loop ---
// C += A[m0.., :] * B[n0.., :]^T with A,B row-major (contraction along rows),
// double-buffered K=64 LDS stages. lA/lB are [2][128][LDP] flat. 8 waves:
// wave (wm,wn) owns a 32x64 sub-block = 2x4 accumulator tiles.
__device__ __forceinline__ void gemm_mainloop(
    const bf16* __restrict__ A, const bf16* __restrict__ B,
    int m0, int n0, int Kdim, bf16* lA, bf16* lB, v8f acc[2][4]) {
  const int tid = threadIdx.x;
  const int wid = tid >> 5;
  const int wm = wid & 3, wn = wid >> 2;

  // per-stage cooperative copy: 128 rows x 64 cols per array, 16B per op,
  // 1024 ops per array / 256 threads = 4 each.
  auto stage_copy = [&](int k0, int buf) {
    #pragma unroll
    for (int i = 0; i < 4; ++i) {
      int c = tid + i * 256;        // 0..1023
      int row = c >> 3, q = c & 7;  // 8 x 8-elem chunks per row
      cp16_g2l(A + (size_t)(m0 + row) * HID + k0 + q * 8,
               lA + (size_t)(buf * 128 + row) * LDP + q * 8);
      cp16_g2l(B + (size_t)(n0 + row) * HID + k0 + q * 8,
               lB + (size_t)(buf * 128 + row) * LDP + q * 8);
    }
  };

  stage_copy(0, 0);
  const int T = Kdim / KSTEP;
  for (int t = 0; t < T; ++t) {
    async_wait0();
    __syncthreads();  // stage t visible; all readers of buf[(t+1)&1] retired
    if (t + 1 < T) stage_copy((t + 1) * KSTEP, (t + 1) & 1);
    const bf16* bA = lA + (size_t)(t & 1) * 128 * LDP;
    const bf16* bB = lB + (size_t)(t & 1) * 128 * LDP;
    #pragma unroll
    for (int ks = 0; ks < KSTEP; ks += 32) {
      v16bf a[2], b[4];
      #pragma unroll
      for (int mi = 0; mi < 2; ++mi)
        a[mi] = frag_a(bA + (size_t)(wm * 32 + mi * 16) * LDP + ks, LDP);
      #pragma unroll
      for (int ni = 0; ni < 4; ++ni)
        b[ni] = frag_b(bB + (size_t)(wn * 64 + ni * 16) * LDP + ks, LDP);
      #pragma unroll
      for (int mi = 0; mi < 2; ++mi)
        #pragma unroll
        for (int ni = 0; ni < 4; ++ni)
          acc[mi][ni] = wmma_bf16(a[mi], b[ni], acc[mi][ni]);
    }
  }
}

// ---------------------------------------------------------------- convert ----
typedef struct __attribute__((aligned(8))) { bf16 x, y, z, w; } bf16x4;

__global__ void cvt_f32_bf16(const float* __restrict__ s, bf16* __restrict__ d, int n4) {
  int i = blockIdx.x * blockDim.x + threadIdx.x;
  if (i < n4) {
    float4 f = ((const float4*)s)[i];
    bf16x4 o;
    o.x = (bf16)f.x; o.y = (bf16)f.y; o.z = (bf16)f.z; o.w = (bf16)f.w;
    ((bf16x4*)d)[i] = o;
  }
}

// ----------------------------------------------- QKV GEMM + bias + RoPE ------
// C[s, o] = sum_h X[s,h] * W[o,h] + bias[o]; each 128-wide o-block is exactly
// one of {q,k,v} for one head (384 = 3*128). RoPE pairs (j, j+32) inside each
// 64-dim half live in accumulator tiles (ni, ni+2) of the same lane.
__global__ __launch_bounds__(256) void qkv_rope_kernel(
    const bf16* __restrict__ X, const bf16* __restrict__ W,
    const float* __restrict__ bias, const int* __restrict__ posids,
    bf16* __restrict__ Qb, bf16* __restrict__ Kb, bf16* __restrict__ Vt) {
  __shared__ bf16 lA[2 * 128 * LDP];
  __shared__ bf16 lB[2 * 128 * LDP];
  const int tid = threadIdx.x;
  const int lane = tid & 31, wid = tid >> 5;
  const int wm = wid & 3, wn = wid >> 2;
  const int n = lane & 15, half = lane >> 4;
  const int m0 = blockIdx.y * 128;
  const int n0 = blockIdx.x * 128;

  v8f acc[2][4];
  #pragma unroll
  for (int i = 0; i < 2; ++i)
    #pragma unroll
    for (int j = 0; j < 4; ++j) acc[i][j] = vzero8();

  gemm_mainloop(X, W, m0, n0, HID, lA, lB, acc);

  // bias
  #pragma unroll
  for (int ni = 0; ni < 4; ++ni) {
    float bv = bias[n0 + wn * 64 + ni * 16 + n];
    #pragma unroll
    for (int mi = 0; mi < 2; ++mi)
      #pragma unroll
      for (int v = 0; v < 8; ++v) acc[mi][ni][v] += bv;
  }

  const int part = blockIdx.x % 3;  // 0=q 1=k 2=v
  const int head = blockIdx.x / 3;

  if (part == 2) {
    // store V transposed: Vt[head][d][s] (elements are 8 consecutive s)
    #pragma unroll
    for (int mi = 0; mi < 2; ++mi)
      #pragma unroll
      for (int ni = 0; ni < 4; ++ni)
        #pragma unroll
        for (int v = 0; v < 8; ++v) {
          int c = wn * 64 + ni * 16 + n;
          int s = m0 + wm * 32 + mi * 16 + v + 8 * half;
          Vt[((size_t)head * HDIM + c) * S_LEN + s] = (bf16)acc[mi][ni][v];
        }
  } else {
    // 2D RoPE: wn==0 -> dims 0..63 with pid, wn==1 -> dims 64..127 with bid.
    const int* pp = (wn == 0) ? posids : (posids + S_LEN);
    // 1/(sqrt(HD)*coeff) then *coeff after mask -> net 1/sqrt(HD), folded into Q.
    const float scale = (part == 0) ? 0.08838834764831845f : 1.0f;
    bf16* dst = ((part == 0) ? Qb : Kb) + (size_t)head * S_LEN * HDIM;
    #pragma unroll
    for (int mi = 0; mi < 2; ++mi)
      #pragma unroll
      for (int v = 0; v < 8; ++v) {
        int s = m0 + wm * 32 + mi * 16 + v + 8 * half;
        float pv = (float)pp[s];
        #pragma unroll
        for (int ni = 0; ni < 2; ++ni) {
          int jj = ni * 16 + n;  // 0..31, pairs with jj+32 (tile ni+2)
          // inv_freq = 10000^(-jj/32) = exp2(-jj * log2(10000)/32)
          float ang = pv * exp2f(-0.41524101186092029f * (float)jj);
          float c_ = __cosf(ang), s_ = __sinf(ang);
          float x1 = acc[mi][ni][v], x2 = acc[mi][ni + 2][v];
          acc[mi][ni][v]     = x1 * c_ - x2 * s_;
          acc[mi][ni + 2][v] = x2 * c_ + x1 * s_;
        }
        #pragma unroll
        for (int ni = 0; ni < 4; ++ni) {
          int d = wn * 64 + ni * 16 + n;
          dst[(size_t)s * HDIM + d] = (bf16)(acc[mi][ni][v] * scale);
        }
      }
  }
}

// ------------------------------------------------------- flash attention -----
__global__ __launch_bounds__(256) void attn_kernel(
    const bf16* __restrict__ Qb, const bf16* __restrict__ Kb,
    const bf16* __restrict__ Vt, bf16* __restrict__ Ctx) {
  __shared__ bf16 lP[8][16][40];  // per-wave P scratch (C-layout -> A-layout)
  const int tid = threadIdx.x, lane = tid & 31, wid = tid >> 5;
  const int n = lane & 15, half = lane >> 4;
  const int head = blockIdx.y;
  const int m0 = blockIdx.x * 128 + wid * 16;  // each wave: 16 query rows
  const bf16* Qh = Qb + (size_t)head * S_LEN * HDIM;
  const bf16* Kh = Kb + (size_t)head * S_LEN * HDIM;
  const bf16* Vh = Vt + (size_t)head * HDIM * S_LEN;

  v16bf qf[4];
  #pragma unroll
  for (int kk = 0; kk < 4; ++kk)
    qf[kk] = frag_a(Qh + (size_t)m0 * HDIM + kk * 32, HDIM);

  v8f o_acc[8];
  #pragma unroll
  for (int i = 0; i < 8; ++i) o_acc[i] = vzero8();
  float m_run[8], l_run[8];
  #pragma unroll
  for (int v = 0; v < 8; ++v) { m_run[v] = -1e30f; l_run[v] = 0.0f; }

  // Per-wave causal bound: last key needed is m0+15, so j0 <= m0 always and
  // no row is ever fully masked in an executed iteration.
  const int jmax = m0 + 16;
  for (int j0 = 0; j0 < jmax; j0 += 32) {
    v8f sc0 = vzero8(), sc1 = vzero8();
    #pragma unroll
    for (int kk = 0; kk < 4; ++kk) {
      v16bf bk0 = frag_b(Kh + (size_t)j0 * HDIM + kk * 32, HDIM);
      v16bf bk1 = frag_b(Kh + (size_t)(j0 + 16) * HDIM + kk * 32, HDIM);
      sc0 = wmma_bf16(qf[kk], bk0, sc0);
      sc1 = wmma_bf16(qf[kk], bk1, sc1);
    }
    float corr[8];
    #pragma unroll
    for (int v = 0; v < 8; ++v) {
      int s = m0 + v + 8 * half;
      if (j0 + n > s)      sc0[v] = -1e30f;   // causal mask (== -10000*coeff path)
      if (j0 + 16 + n > s) sc1[v] = -1e30f;
      float mx = fmaxf(sc0[v], sc1[v]);
      mx = fmaxf(mx, __shfl_xor(mx, 1, 32));
      mx = fmaxf(mx, __shfl_xor(mx, 2, 32));
      mx = fmaxf(mx, __shfl_xor(mx, 4, 32));
      mx = fmaxf(mx, __shfl_xor(mx, 8, 32));
      float mnew = fmaxf(m_run[v], mx);
      corr[v] = __expf(m_run[v] - mnew);
      float p0 = __expf(sc0[v] - mnew);
      float p1 = __expf(sc1[v] - mnew);
      float rs = p0 + p1;
      rs += __shfl_xor(rs, 1, 32);
      rs += __shfl_xor(rs, 2, 32);
      rs += __shfl_xor(rs, 4, 32);
      rs += __shfl_xor(rs, 8, 32);
      l_run[v] = l_run[v] * corr[v] + rs;
      m_run[v] = mnew;
      lP[wid][v + 8 * half][n]      = (bf16)p0;
      lP[wid][v + 8 * half][16 + n] = (bf16)p1;
    }
    #pragma unroll
    for (int dt = 0; dt < 8; ++dt)
      #pragma unroll
      for (int v = 0; v < 8; ++v) o_acc[dt][v] *= corr[v];
    v16bf ap = frag_a(&lP[wid][0][0], 40);
    #pragma unroll
    for (int dt = 0; dt < 8; ++dt) {
      v16bf bv = frag_b(Vh + (size_t)(dt * 16) * S_LEN + j0, S_LEN);
      o_acc[dt] = wmma_bf16(ap, bv, o_acc[dt]);
    }
  }

  #pragma unroll
  for (int v = 0; v < 8; ++v) {
    float inv = 1.0f / l_run[v];
    int s = m0 + v + 8 * half;
    #pragma unroll
    for (int dt = 0; dt < 8; ++dt) {
      int d = dt * 16 + n;
      Ctx[(size_t)s * HID + head * HDIM + d] = (bf16)(o_acc[dt][v] * inv);
    }
  }
}

// ------------------------------------------------------ output projection ----
__global__ __launch_bounds__(256) void out_proj_kernel(
    const bf16* __restrict__ A, const bf16* __restrict__ W,
    const float* __restrict__ bias, float* __restrict__ out) {
  __shared__ bf16 lA[2 * 128 * LDP];
  __shared__ bf16 lB[2 * 128 * LDP];
  const int tid = threadIdx.x;
  const int lane = tid & 31, wid = tid >> 5;
  const int wm = wid & 3, wn = wid >> 2;
  const int n = lane & 15, half = lane >> 4;
  const int m0 = blockIdx.y * 128;
  const int n0 = blockIdx.x * 128;

  v8f acc[2][4];
  #pragma unroll
  for (int i = 0; i < 2; ++i)
    #pragma unroll
    for (int j = 0; j < 4; ++j) acc[i][j] = vzero8();

  gemm_mainloop(A, W, m0, n0, HID, lA, lB, acc);

  #pragma unroll
  for (int ni = 0; ni < 4; ++ni) {
    int col = n0 + wn * 64 + ni * 16 + n;
    float bv = bias[col];
    #pragma unroll
    for (int mi = 0; mi < 2; ++mi)
      #pragma unroll
      for (int v = 0; v < 8; ++v) {
        int s = m0 + wm * 32 + mi * 16 + v + 8 * half;
        out[(size_t)s * HID + col] = acc[mi][ni][v] + bv;
      }
  }
}

// ------------------------------------------------------------------ launch ---
extern "C" void kernel_launch(void* const* d_in, const int* in_sizes, int n_in,
                              void* d_out, int out_size, void* d_ws, size_t ws_size,
                              hipStream_t stream) {
  const float* hidden = (const float*)d_in[0];
  const int*   posids = (const int*)d_in[1];  // [1,2,S]: pid = [0..S), bid = [S..2S)
  // d_in[2] attention_mask: causal, handled structurally. d_in[3] layer_id: cancels.
  const float* Wqkv = (const float*)d_in[4];
  const float* bqkv = (const float*)d_in[5];
  const float* Wd   = (const float*)d_in[6];
  const float* bd   = (const float*)d_in[7];
  float* out = (float*)d_out;

  char* ws = (char*)d_ws;
  const size_t MB = 1024 * 1024;
  bf16* Xbf   = (bf16*)(ws);             // 16 MiB  [S][H]
  bf16* Wqkvb = (bf16*)(ws + 16 * MB);   // 96 MiB  [3H][H]
  bf16* Wdb   = (bf16*)(ws + 112 * MB);  // 32 MiB  [H][H]
  bf16* Qb    = (bf16*)(ws + 144 * MB);  // 16 MiB  [NH][S][HD]
  bf16* Kb    = (bf16*)(ws + 160 * MB);  // 16 MiB  [NH][S][HD]
  bf16* Vt    = (bf16*)(ws + 176 * MB);  // 16 MiB  [NH][HD][S] (transposed)
  bf16* Ctx   = (bf16*)(ws + 192 * MB);  // 16 MiB  [S][H]; total 208 MiB

  {
    int n4 = (int)((size_t)S_LEN * HID / 4);
    cvt_f32_bf16<<<(n4 + 255) / 256, 256, 0, stream>>>(hidden, Xbf, n4);
    n4 = (int)((size_t)QKVN * HID / 4);
    cvt_f32_bf16<<<(n4 + 255) / 256, 256, 0, stream>>>(Wqkv, Wqkvb, n4);
    n4 = (int)((size_t)HID * HID / 4);
    cvt_f32_bf16<<<(n4 + 255) / 256, 256, 0, stream>>>(Wd, Wdb, n4);
  }

  qkv_rope_kernel<<<dim3(QKVN / 128, S_LEN / 128), 256, 0, stream>>>(
      Xbf, Wqkvb, bqkv, posids, Qb, Kb, Vt);

  attn_kernel<<<dim3(S_LEN / 128, NHEAD), 256, 0, stream>>>(Qb, Kb, Vt, Ctx);

  out_proj_kernel<<<dim3(HID / 128, S_LEN / 128), 256, 0, stream>>>(Ctx, Wdb, bd, out);
}
